// MelSpectrogramONNXWrapper_6811818132048
// MI455X (gfx1250) — compile-verified, matrix operand order checked by reading, not run on hardware
//
#include <hip/hip_runtime.h>
#include <hip/hip_bf16.h>

typedef __attribute__((ext_vector_type(16))) _Float16 v16h;
typedef __attribute__((ext_vector_type(8)))  float    v8f;

// ---------------- problem constants ----------------
#define BATCH     16
#define T_AUDIO   320000
#define N_FFT     1024
#define HOP       256
#define PAD       512
#define NFRAMES   1251                 // (320000+1024-1024)/256 + 1
#define M_ROWS    (BATCH * NFRAMES)    // 20016 = 1251 * 16  (exact M tiles!)
#define MT        1251                 // M tiles of 16
#define MG        626                  // groups of 2 mTiles (last group clamped)
#define KT1       32                   // 1024 / 32 K-steps for DFT GEMM
#define N_STFT    513
#define NCOL      528                  // 513 padded to 33 tiles of 16
#define NPAIR     33                   // real/imag tile pairs
#define NT1       66                   // total B1 N tiles (real 0..32, imag 33..65)
#define KT2       17                   // 544 / 32 K-steps for mel GEMM (528 padded)
#define N_MELS    80
#define NT2       5                    // 80 / 16

// ---------------- workspace layout (halves) ----------------
#define A1_OFF   0ull
#define A1_SZ    (1251ull*32*512)
#define B1_OFF   (A1_OFF + A1_SZ)
#define B1_SZ    (66ull*32*512)
#define MAG_OFF  (B1_OFF + B1_SZ)
#define MAG_SZ   (20016ull*528)
#define A2_OFF   (MAG_OFF + MAG_SZ)
#define A2_SZ    (1251ull*17*512)
#define B2_OFF   (A2_OFF + A2_SZ)

// A/B fragment K index for (vgpr v, half h, lane): ISA 05_wmma 16-bit 16x32 layout
__device__ __forceinline__ int frag_k(int e, int lane) {
    int v = e >> 1, h = e & 1;
    return 2 * (v & 3) + h + ((v >> 2) << 4) + ((lane >> 4) << 3);
}

__device__ __forceinline__ v8f wmma_f16(v16h a, v16h b, v8f c) {
    return __builtin_amdgcn_wmma_f32_16x16x32_f16(false, a, false, b,
                                                  (short)0, c, false, false);
}

// ---------------- kernel 1: windowed frames -> A1 fragments (f16) ----------------
__global__ __launch_bounds__(256)
void pack_frames(const float* __restrict__ wav, const float* __restrict__ window,
                 _Float16* __restrict__ a1) {
    int id = blockIdx.x * blockDim.x + threadIdx.x;     // one thread = one lane-frag
    const int total = MT * KT1 * 32;
    if (id >= total) return;
    int lane  = id & 31;
    int t     = id >> 5;
    int kTile = t % KT1;
    int mTile = t / KT1;
    int row   = lane & 15;
    int m     = mTile * 16 + row;
    int b     = m / NFRAMES;
    int fr    = m % NFRAMES;
    v16h frag;
#pragma unroll
    for (int e = 0; e < 16; ++e) {
        int K = kTile * 32 + frag_k(e, lane);
        int p = fr * HOP + K - PAD;                      // reflect padding
        if (p < 0) p = -p;
        if (p >= T_AUDIO) p = 2 * (T_AUDIO - 1) - p;
        frag[e] = (_Float16)(wav[b * T_AUDIO + p] * window[K]);
    }
    *(v16h*)(a1 + (size_t)id * 16) = frag;
}

// ---------------- kernel 2: DFT basis (real||imag) -> B1 fragments ----------------
__global__ __launch_bounds__(256)
void pack_dft(const float* __restrict__ re_t, const float* __restrict__ im_t,
              _Float16* __restrict__ b1) {
    int id = blockIdx.x * blockDim.x + threadIdx.x;
    const int total = NT1 * KT1 * 32;
    if (id >= total) return;
    int lane  = id & 31;
    int t     = id >> 5;
    int kTile = t % KT1;
    int nTile = t / KT1;
    int col   = nTile * 16 + (lane & 15);                // 0..1055
    v16h frag;
#pragma unroll
    for (int e = 0; e < 16; ++e) {
        int K = kTile * 32 + frag_k(e, lane);
        float v = 0.0f;
        if (col < NCOL) { if (col < N_STFT)        v = re_t[K * N_STFT + col]; }
        else            { int c = col - NCOL; if (c < N_STFT) v = im_t[K * N_STFT + c]; }
        frag[e] = (_Float16)v;
    }
    *(v16h*)(b1 + (size_t)id * 16) = frag;
}

// ---------------- kernel 3: mel fbanks -> B2 fragments ----------------
__global__ __launch_bounds__(256)
void pack_mel(const float* __restrict__ fb, _Float16* __restrict__ b2) {
    int id = blockIdx.x * blockDim.x + threadIdx.x;
    const int total = NT2 * KT2 * 32;
    if (id >= total) return;
    int lane  = id & 31;
    int t     = id >> 5;
    int kTile = t % KT2;
    int nTile = t / KT2;
    int N     = nTile * 16 + (lane & 15);                // < 80
    v16h frag;
#pragma unroll
    for (int e = 0; e < 16; ++e) {
        int K = kTile * 32 + frag_k(e, lane);
        frag[e] = (_Float16)((K < N_STFT) ? fb[K * N_MELS + N] : 0.0f);
    }
    *(v16h*)(b2 + (size_t)id * 16) = frag;
}

// ---- kernel 4: DFT GEMM, 2 mTiles x (real,imag) pair per wave + fused magnitude ----
__global__ __launch_bounds__(128)
void dft_gemm_mag(const _Float16* __restrict__ a1, const _Float16* __restrict__ b1,
                  _Float16* __restrict__ mag) {
    int lane = threadIdx.x & 31;
    int wave = threadIdx.x >> 5;
    int g    = blockIdx.x * 4 + wave;                    // wave-uniform group
    if (g >= MG) return;
    int m0 = g * 2;
    int m1 = (m0 + 1 < MT) ? (m0 + 1) : m0;              // clamp last odd tile
    int j  = blockIdx.y;                                 // 0..32 column-pair

    const v16h* A0 = (const v16h*)a1 + ((size_t)m0        * KT1 * 32 + lane);
    const v16h* A1 = (const v16h*)a1 + ((size_t)m1        * KT1 * 32 + lane);
    const v16h* Br = (const v16h*)b1 + ((size_t)j         * KT1 * 32 + lane);
    const v16h* Bi = (const v16h*)b1 + ((size_t)(j+NPAIR) * KT1 * 32 + lane);

    v8f aR0 = {}, aI0 = {}, aR1 = {}, aI1 = {};
#pragma unroll 2
    for (int k = 0; k < KT1; ++k) {
        v16h a0 = A0[k * 32];
        v16h a1v = A1[k * 32];
        v16h br = Br[k * 32];
        v16h bi = Bi[k * 32];
        aR0 = wmma_f16(a0,  br, aR0);
        aI0 = wmma_f16(a0,  bi, aI0);
        aR1 = wmma_f16(a1v, br, aR1);
        aI1 = wmma_f16(a1v, bi, aI1);
    }
    // fused magnitude: C/D layout -> lane%16 = col, vgpr r + 8*(lane/16) = row
    int col = j * 16 + (lane & 15);
#pragma unroll
    for (int r = 0; r < 8; ++r) {
        int m = m0 * 16 + r + ((lane >> 4) << 3);
        float re = aR0[r], im = aI0[r];
        mag[(size_t)m * NCOL + col] = (_Float16)sqrtf(re * re + im * im);
    }
    if (m1 != m0) {
#pragma unroll
        for (int r = 0; r < 8; ++r) {
            int m = m1 * 16 + r + ((lane >> 4) << 3);
            float re = aR1[r], im = aI1[r];
            mag[(size_t)m * NCOL + col] = (_Float16)sqrtf(re * re + im * im);
        }
    }
}

// ---------------- kernel 5: magnitudes -> A2 fragments ----------------
__global__ __launch_bounds__(256)
void pack_mag(const _Float16* __restrict__ mag, _Float16* __restrict__ a2) {
    int id = blockIdx.x * blockDim.x + threadIdx.x;
    const int total = MT * KT2 * 32;
    if (id >= total) return;
    int lane  = id & 31;
    int t     = id >> 5;
    int kTile = t % KT2;
    int mTile = t / KT2;
    int m     = mTile * 16 + (lane & 15);
    v16h frag;
#pragma unroll
    for (int e = 0; e < 16; ++e) {
        int K = kTile * 32 + frag_k(e, lane);
        frag[e] = (K < NCOL) ? mag[(size_t)m * NCOL + K] : (_Float16)0.0f;
    }
    *(v16h*)(a2 + (size_t)id * 16) = frag;
}

// ---------------- kernel 6: mel GEMM (2 mTiles/wave) + transposed store ----------------
__global__ __launch_bounds__(128)
void mel_gemm(const _Float16* __restrict__ a2, const _Float16* __restrict__ b2,
              float* __restrict__ out) {
    int lane = threadIdx.x & 31;
    int wave = threadIdx.x >> 5;
    int g    = blockIdx.x * 4 + wave;                    // wave-uniform group
    if (g >= MG) return;
    int m0 = g * 2;
    int m1 = (m0 + 1 < MT) ? (m0 + 1) : m0;

    const v16h* A0 = (const v16h*)a2 + ((size_t)m0 * KT2 * 32 + lane);
    const v16h* A1 = (const v16h*)a2 + ((size_t)m1 * KT2 * 32 + lane);
    const v16h* Bf = (const v16h*)b2 + lane;

    v8f acc0[NT2] = {};
    v8f acc1[NT2] = {};
#pragma unroll 1
    for (int k = 0; k < KT2; ++k) {
        v16h a0 = A0[k * 32];
        v16h a1v = A1[k * 32];
#pragma unroll
        for (int n = 0; n < NT2; ++n) {
            v16h b = Bf[((size_t)n * KT2 + k) * 32];
            acc0[n] = wmma_f16(a0,  b, acc0[n]);
            acc1[n] = wmma_f16(a1v, b, acc1[n]);
        }
    }
    // epilogue: out[b][mel][frame]
#pragma unroll
    for (int n = 0; n < NT2; ++n) {
        int mel = n * 16 + (lane & 15);
#pragma unroll
        for (int r = 0; r < 8; ++r) {
            int m  = m0 * 16 + r + ((lane >> 4) << 3);
            int b  = m / NFRAMES;
            int fr = m % NFRAMES;
            out[((size_t)b * N_MELS + mel) * NFRAMES + fr] = acc0[n][r];
        }
    }
    if (m1 != m0) {
#pragma unroll
        for (int n = 0; n < NT2; ++n) {
            int mel = n * 16 + (lane & 15);
#pragma unroll
            for (int r = 0; r < 8; ++r) {
                int m  = m1 * 16 + r + ((lane >> 4) << 3);
                int b  = m / NFRAMES;
                int fr = m % NFRAMES;
                out[((size_t)b * N_MELS + mel) * NFRAMES + fr] = acc1[n][r];
            }
        }
    }
}

// ---------------- launch ----------------
extern "C" void kernel_launch(void* const* d_in, const int* in_sizes, int n_in,
                              void* d_out, int out_size, void* d_ws, size_t ws_size,
                              hipStream_t stream) {
    const float* wav    = (const float*)d_in[0];   // (16,1,320000)
    const float* window = (const float*)d_in[1];   // (1024)
    const float* fb     = (const float*)d_in[2];   // (513,80)
    const float* re_t   = (const float*)d_in[3];   // (1024,513)
    const float* im_t   = (const float*)d_in[4];   // (1024,513)
    float* out = (float*)d_out;                    // (16,80,1251)

    _Float16* ws  = (_Float16*)d_ws;
    _Float16* a1  = ws + A1_OFF;
    _Float16* b1  = ws + B1_OFF;
    _Float16* mag = ws + MAG_OFF;
    _Float16* a2  = ws + A2_OFF;
    _Float16* b2  = ws + B2_OFF;

    // pack stages (independent)
    {   int total = MT * KT1 * 32;
        pack_frames<<<(total + 255) / 256, 256, 0, stream>>>(wav, window, a1); }
    {   int total = NT1 * KT1 * 32;
        pack_dft<<<(total + 255) / 256, 256, 0, stream>>>(re_t, im_t, b1); }
    {   int total = NT2 * KT2 * 32;
        pack_mel<<<(total + 255) / 256, 256, 0, stream>>>(fb, b2); }

    // DFT GEMM + magnitude: 626 groups of 2 mTiles x 33 column pairs, 4 waves / block
    dft_gemm_mag<<<dim3((MG + 3) / 4, NPAIR), 128, 0, stream>>>(a1, b1, mag);

    // repack magnitudes into A fragments
    {   int total = MT * KT2 * 32;
        pack_mag<<<(total + 255) / 256, 256, 0, stream>>>(mag, a2); }

    // mel GEMM + transposed write
    mel_gemm<<<(MG + 3) / 4, 128, 0, stream>>>(a2, b2, out);
}
